// GroupedQueryAttention_20925080666571
// MI455X (gfx1250) — matrix operand order repaired and checked by reading, not compiled, not run on hardware
//
#include <hip/hip_runtime.h>
#include <hip/hip_bf16.h>
#include <math.h>

// ---------------------------------------------------------------------------
// GQA block for MI455X (gfx1250, wave32).
// bf16 WMMA (v_wmma_f32_16x16x32_bf16) everywhere; f32 accumulate.
// Round 3: ping-pong (2x-unrolled) GEMM pipeline -> no inter-iteration
// register rotation copies; loads land in the WMMA operand registers.
// ---------------------------------------------------------------------------

typedef __bf16 bf16;
typedef __attribute__((ext_vector_type(16))) __bf16 v16bf;
typedef __attribute__((ext_vector_type(8)))  __bf16 v8bf;
typedef __attribute__((ext_vector_type(8)))  float  v8f;

#define S_LEN    2048
#define D_MODEL  4096
#define KV_DIM   1024
#define N_HEADS  32
#define N_GROUPS 8
#define HEAD_DIM 128

// Assemble a 16-element bf16 fragment from two contiguous 8-element chunks.
// Callers pass p1 = p0 + 8: each lane half holds 16 contiguous K (k = 16*(lane>>4)
// + s), A with lane = row and B (stored transposed) with lane = column. Not the
// ISA table K order, but A and B use the same one, so the product is right.
__device__ __forceinline__ v16bf load_frag16(const bf16* p0, const bf16* p1) {
    v8bf lo = *(const v8bf*)p0;
    v8bf hi = *(const v8bf*)p1;
    v16bf a;
#pragma unroll
    for (int i = 0; i < 8; ++i) { a[i] = lo[i]; a[i + 8] = hi[i]; }
    return a;
}

__device__ __forceinline__ v8f wmma_bf16(v16bf a, v16bf b, v8f c) {
    return __builtin_amdgcn_wmma_f32_16x16x32_bf16(
        /*neg_a=*/false, a, /*neg_b=*/false, b,
        /*c_mod=*/(short)0, c, /*reuse_a=*/false, /*reuse_b=*/false);
}

// ---------------------------------------------------------------------------
// f32 -> bf16 elementwise convert
// ---------------------------------------------------------------------------
__global__ void cvt_bf16_kernel(const float* __restrict__ src,
                                bf16* __restrict__ dst, int n) {
    int i = blockIdx.x * blockDim.x + threadIdx.x;
    if (i < n) dst[i] = (bf16)src[i];
}

// f32 W[K][N] -> bf16 Wt[N][K] (coalesced read, scattered write)
__global__ void cvt_t_bf16_kernel(const float* __restrict__ W,
                                  bf16* __restrict__ Wt, int K, int N) {
    int i = blockIdx.x * blockDim.x + threadIdx.x;
    if (i >= K * N) return;
    int n = i % N;
    int k = i / N;
    Wt[(size_t)n * K + k] = (bf16)W[i];
}

// f32 V[S][KV_DIM] -> bf16 Vt[KV_DIM][S]
__global__ void v_transpose_kernel(const float* __restrict__ Vf,
                                   bf16* __restrict__ Vt) {
    int i = blockIdx.x * blockDim.x + threadIdx.x;
    int s = i / KV_DIM;
    int c = i % KV_DIM;
    Vt[(size_t)c * S_LEN + s] = (bf16)Vf[i];
}

// RoPE on f32 projections, emit bf16. One thread per (s, head, freq) pair:
// rotates elements d and d+64 of the 128-wide head.
__global__ void rope_cvt_kernel(const float* __restrict__ src,
                                bf16* __restrict__ dst, int heads) {
    int i = blockIdx.x * blockDim.x + threadIdx.x;
    int total = S_LEN * heads * (HEAD_DIM / 2);
    if (i >= total) return;
    int f = i & 63;                 // freq index 0..63
    int h = (i >> 6) % heads;
    int s = i / (heads * 64);
    int dim = heads * HEAD_DIM;
    float inv = __powf(10000.0f, -(float)f * (1.0f / 64.0f));
    float ang = (float)s * inv;
    float c = __cosf(ang), sn = __sinf(ang);
    size_t base = (size_t)s * dim + (size_t)h * HEAD_DIM;
    float x1 = src[base + f];
    float x2 = src[base + 64 + f];
    dst[base + f]      = (bf16)(x1 * c - x2 * sn);
    dst[base + 64 + f] = (bf16)(x2 * c + x1 * sn);
}

// ---------------------------------------------------------------------------
// C_f32[M][N] = A_bf16[M][K] @ Bt_bf16[N][K]^T
// One wave -> 16x64 C tile (4 accumulators), K stepped by 32.
// Ping-pong pipelined: two named fragment buffers alternate roles in a
// 2x-unrolled loop, so loads land directly in WMMA operand registers and one
// full 10-load tile-set is always in flight (partial s_wait_loadcnt).
// Requires K % 64 == 0 (true for all uses here: K = 4096).
// Guards are wave-uniform so EXEC is all-ones at every WMMA.
// ---------------------------------------------------------------------------
__global__ void __launch_bounds__(256)
gemm_bf16_kernel(const bf16* __restrict__ A, const bf16* __restrict__ Bt,
                 float* __restrict__ C, int M, int N, int K) {
    const int lane  = threadIdx.x & 31;
    const int wid   = threadIdx.x >> 5;
    const int gw    = blockIdx.x * 8 + wid;
    const int ntil  = N >> 6;
    const int m0    = (gw / ntil) << 4;
    const int n0    = (gw % ntil) << 6;
    if (m0 >= M) return;                      // wave-uniform
    const int l15   = lane & 15;
    const int kHalf = lane >> 4;

    v8f acc[4];
#pragma unroll
    for (int j = 0; j < 4; ++j) acc[j] = (v8f){0.f,0.f,0.f,0.f,0.f,0.f,0.f,0.f};

    const bf16* aRow = A + (size_t)(m0 + l15) * K + kHalf * 16;
    const bf16* bRow[4];
#pragma unroll
    for (int j = 0; j < 4; ++j)
        bRow[j] = Bt + (size_t)(n0 + j * 16 + l15) * K + kHalf * 16;

    auto loadSet = [&](v16bf& aF, v16bf (&bF)[4], int t) {
        const int off = t << 5;
        aF = load_frag16(aRow + off, aRow + off + 8);
#pragma unroll
        for (int j = 0; j < 4; ++j)
            bF[j] = load_frag16(bRow[j] + off, bRow[j] + off + 8);
    };
    auto mmaSet = [&](const v16bf& aF, const v16bf (&bF)[4]) {
#pragma unroll
        for (int j = 0; j < 4; ++j) acc[j] = wmma_bf16(aF, bF[j], acc[j]);
    };

    const int ntiles = K >> 5;                // even (K % 64 == 0)
    v16bf aA, aB, bA[4], bB[4];
    loadSet(aA, bA, 0);
    loadSet(aB, bB, 1);

    for (int t = 0; t + 3 < ntiles; t += 2) {
        __builtin_prefetch(aRow + ((t + 8) << 5), 0, 3);
        mmaSet(aA, bA);                       // consume tile t
        loadSet(aA, bA, t + 2);               // refill A buffer
        mmaSet(aB, bB);                       // consume tile t+1
        loadSet(aB, bB, t + 3);               // refill B buffer
    }
    mmaSet(aA, bA);                           // tile ntiles-2
    mmaSet(aB, bB);                           // tile ntiles-1

#pragma unroll
    for (int j = 0; j < 4; ++j)
#pragma unroll
        for (int r = 0; r < 8; ++r)
            C[(size_t)(m0 + kHalf * 8 + r) * N + n0 + j * 16 + l15] = acc[j][r];
}

// ---------------------------------------------------------------------------
// Causal flash attention. One wave per (head, 16-query tile); 32 keys/iter.
// Q,K bf16 row-major (post-RoPE); V bf16 transposed [d][s]; ctx bf16 out.
// K/V fragments are batch-loaded (one clause) before the WMMA bursts so the
// load latency drains under the matrix pipe; V loads overlap the softmax
// rescale VALU work. Per-wave LDS stages P (C-fragment -> A-fragment layout);
// per-wave LDS ops are issued in order, so only a compiler wave barrier is
// needed (all 8 waves of a block share qt and hence the same causal trip
// count).
// ---------------------------------------------------------------------------
__global__ void __launch_bounds__(256)
attn_kernel(const bf16* __restrict__ Qb, const bf16* __restrict__ Kb,
            const bf16* __restrict__ Vt, bf16* __restrict__ ctx) {
    __shared__ bf16 sP[8][16 * 32];           // 1 KiB per wave
    const int lane  = threadIdx.x & 31;
    const int wid   = threadIdx.x >> 5;
    const int gw    = blockIdx.x * 8 + wid;   // 0..4095
    const int h     = gw & (N_HEADS - 1);
    const int qt    = gw >> 5;                // query tile 0..127
    const int g     = h >> 2;                 // KV group (gs = 4)
    const int m0    = qt << 4;
    const int l15   = lane & 15;
    const int kHalf = lane >> 4;
    bf16* pbuf = &sP[wid][0];

    // Preload the four 16x32 Q fragments covering hd = 128.
    v16bf qf[4];
    const bf16* qRow = Qb + (size_t)(m0 + l15) * D_MODEL + h * HEAD_DIM;
#pragma unroll
    for (int dc = 0; dc < 4; ++dc)
        qf[dc] = load_frag16(qRow + dc * 32 + kHalf * 16,
                             qRow + dc * 32 + kHalf * 16 + 8);

    float mrow[8], lrow[8];
    v8f o[8];
#pragma unroll
    for (int r = 0; r < 8; ++r) { mrow[r] = -1e30f; lrow[r] = 0.f; }
#pragma unroll
    for (int u = 0; u < 8; ++u) o[u] = (v8f){0.f,0.f,0.f,0.f,0.f,0.f,0.f,0.f};

    const float scale = 0.08838834764831845f; // 1/sqrt(128)
    const int nTiles = (qt >> 1) + 1;         // 32-key tiles covering 0..m0+15

    for (int t = 0; t < nTiles; ++t) {
        const int n0 = t << 5;

        // batch-load all 8 K fragments (two 16-key column tiles x 4 d-chunks)
        v16bf kb0[4], kb1[4];
#pragma unroll
        for (int dc = 0; dc < 4; ++dc) {
            const bf16* k0p = Kb + (size_t)(n0 + l15) * KV_DIM
                                 + g * HEAD_DIM + dc * 32 + kHalf * 16;
            kb0[dc] = load_frag16(k0p, k0p + 8);
            const bf16* k1p = k0p + (size_t)16 * KV_DIM;
            kb1[dc] = load_frag16(k1p, k1p + 8);
        }
        v8f s0 = (v8f){0.f,0.f,0.f,0.f,0.f,0.f,0.f,0.f};
        v8f s1 = s0;
#pragma unroll
        for (int dc = 0; dc < 4; ++dc) s0 = wmma_bf16(qf[dc], kb0[dc], s0);
#pragma unroll
        for (int dc = 0; dc < 4; ++dc) s1 = wmma_bf16(qf[dc], kb1[dc], s1);

        // Online softmax. C fragment: row = kHalf*8 + r, col = l15;
        // each row lives in one 16-lane half -> width-16 xor reductions.
        float alpha[8];
#pragma unroll
        for (int r = 0; r < 8; ++r) {
            const int qrow = m0 + kHalf * 8 + r;
            float v0 = (n0 + l15      <= qrow) ? s0[r] * scale : -1e30f;
            float v1 = (n0 + 16 + l15 <= qrow) ? s1[r] * scale : -1e30f;
            float mx = fmaxf(v0, v1);
            for (int msk = 1; msk < 16; msk <<= 1)
                mx = fmaxf(mx, __shfl_xor(mx, msk, 16));
            float mnew = fmaxf(mrow[r], mx);
            float a  = __expf(mrow[r] - mnew);
            float p0 = __expf(v0 - mnew);
            float p1 = __expf(v1 - mnew);
            float rs = p0 + p1;
            for (int msk = 1; msk < 16; msk <<= 1)
                rs += __shfl_xor(rs, msk, 16);
            lrow[r] = lrow[r] * a + rs;
            mrow[r] = mnew;
            alpha[r] = a;
            const int Mr = kHalf * 8 + r;
            pbuf[Mr * 32 + l15]      = (bf16)p0;   // ds_store_b16
            pbuf[Mr * 32 + 16 + l15] = (bf16)p1;
        }
        __builtin_amdgcn_wave_barrier();

        // batch-load all 8 V fragments while rescaling the running context
        v16bf bv[8];
#pragma unroll
        for (int u = 0; u < 8; ++u) {
            const bf16* vp = Vt + (size_t)(g * HEAD_DIM + u * 16 + l15) * S_LEN
                                + n0 + kHalf * 16;
            bv[u] = load_frag16(vp, vp + 8);
        }
#pragma unroll
        for (int u = 0; u < 8; ++u)
#pragma unroll
            for (int r = 0; r < 8; ++r) o[u][r] = o[u][r] * alpha[r];

        // reload P as a 16x32 A fragment (ds_load_b128 x2)
        const bf16* pRow = pbuf + l15 * 32;
    v16bf pf = load_frag16(pRow + kHalf * 16, pRow + kHalf * 16 + 8);

        // O += P(16x32) @ V(32keys x 16d) for each of 8 d-tiles
#pragma unroll
        for (int u = 0; u < 8; ++u)
            o[u] = wmma_bf16(pf, bv[u], o[u]);
    }

    // normalize + store ctx (bf16, row-major [S][D])
#pragma unroll
    for (int u = 0; u < 8; ++u)
#pragma unroll
        for (int r = 0; r < 8; ++r) {
            float val = o[u][r] / lrow[r];
            ctx[(size_t)(m0 + kHalf * 8 + r) * D_MODEL
                + h * HEAD_DIM + u * 16 + l15] = (bf16)val;
        }
}

// ---------------------------------------------------------------------------
// Host-side orchestration (stream-ordered; graph-capture safe).
// Workspace layout (MiB offsets), later stages alias spent regions:
//   [  0.. 16) xb          bf16 x
//   [ 16.. 48) Wqt         -> reused as Qb after Q projection
//   [ 48.. 56) Wkt         -> reused as Kb
//   [ 56.. 64) Wvt         -> reused as Vt
//   [ 64.. 96) Wot
//   [ 96..128) Qf (f32)    -> reused as ctx (bf16)
//   [128..136) Kf (f32)
//   [136..144) Vf (f32)
// Total: 144 MiB.
// ---------------------------------------------------------------------------
extern "C" void kernel_launch(void* const* d_in, const int* in_sizes, int n_in,
                              void* d_out, int out_size, void* d_ws, size_t ws_size,
                              hipStream_t stream) {
    (void)in_sizes; (void)n_in; (void)out_size; (void)ws_size;
    const float* x  = (const float*)d_in[0];
    const float* Wq = (const float*)d_in[1];
    const float* Wk = (const float*)d_in[2];
    const float* Wv = (const float*)d_in[3];
    const float* Wo = (const float*)d_in[4];

    char* ws = (char*)d_ws;
    const size_t MB = 1024u * 1024u;
    bf16*  xb   = (bf16*)(ws + 0 * MB);
    bf16*  Wqt  = (bf16*)(ws + 16 * MB);
    bf16*  Wkt  = (bf16*)(ws + 48 * MB);
    bf16*  Wvt  = (bf16*)(ws + 56 * MB);
    bf16*  Wot  = (bf16*)(ws + 64 * MB);
    float* Qf   = (float*)(ws + 96 * MB);
    float* Kf   = (float*)(ws + 128 * MB);
    float* Vf   = (float*)(ws + 136 * MB);
    bf16*  Qb   = (bf16*)(ws + 16 * MB);   // aliases Wqt (dead after Q proj)
    bf16*  Kb   = (bf16*)(ws + 48 * MB);   // aliases Wkt
    bf16*  Vt   = (bf16*)(ws + 56 * MB);   // aliases Wvt
    bf16*  ctxb = (bf16*)(ws + 96 * MB);   // aliases Qf (dead after RoPE)
    float* outp = (float*)d_out;

    const dim3 blk(256);

    // 1) bf16 conversions / weight transposes
    cvt_bf16_kernel<<<(S_LEN * D_MODEL) / 256, blk, 0, stream>>>(x, xb, S_LEN * D_MODEL);
    cvt_t_bf16_kernel<<<(D_MODEL * D_MODEL) / 256, blk, 0, stream>>>(Wq, Wqt, D_MODEL, D_MODEL);
    cvt_t_bf16_kernel<<<(D_MODEL * KV_DIM) / 256, blk, 0, stream>>>(Wk, Wkt, D_MODEL, KV_DIM);
    cvt_t_bf16_kernel<<<(D_MODEL * KV_DIM) / 256, blk, 0, stream>>>(Wv, Wvt, D_MODEL, KV_DIM);
    cvt_t_bf16_kernel<<<(D_MODEL * D_MODEL) / 256, blk, 0, stream>>>(Wo, Wot, D_MODEL, D_MODEL);

    // 2) projections (WMMA GEMMs)
    gemm_bf16_kernel<<<(S_LEN / 16) * (D_MODEL / 64) / 8, blk, 0, stream>>>(
        xb, Wqt, Qf, S_LEN, D_MODEL, D_MODEL);
    gemm_bf16_kernel<<<(S_LEN / 16) * (KV_DIM / 64) / 8, blk, 0, stream>>>(
        xb, Wkt, Kf, S_LEN, KV_DIM, D_MODEL);
    gemm_bf16_kernel<<<(S_LEN / 16) * (KV_DIM / 64) / 8, blk, 0, stream>>>(
        xb, Wvt, Vf, S_LEN, KV_DIM, D_MODEL);

    // 3) RoPE (Q, K) and V transpose, all emitting bf16
    rope_cvt_kernel<<<(S_LEN * N_HEADS * 64) / 256, blk, 0, stream>>>(Qf, Qb, N_HEADS);
    rope_cvt_kernel<<<(S_LEN * N_GROUPS * 64) / 256, blk, 0, stream>>>(Kf, Kb, N_GROUPS);
    v_transpose_kernel<<<(S_LEN * KV_DIM) / 256, blk, 0, stream>>>(Vf, Vt);

    // 4) causal flash attention (WMMA scores + WMMA P.V)
    attn_kernel<<<(N_HEADS * (S_LEN / 16)) / 8, blk, 0, stream>>>(Qb, Kb, Vt, ctxb);

    // 5) output projection straight into d_out (f32)
    gemm_bf16_kernel<<<(S_LEN / 16) * (D_MODEL / 64) / 8, blk, 0, stream>>>(
        ctxb, Wot, outp, S_LEN, D_MODEL, D_MODEL);
}
